// NGCFModel_22316650070695
// MI455X (gfx1250) — matrix operand order repaired and checked by reading, not compiled, not run on hardware
//
#include <hip/hip_runtime.h>
#include <hip/hip_bf16.h>

#define U_CNT 100000
#define I_CNT 200000
#define N_CNT 300000
#define NNZ_CNT 4000000
#define B_CNT 4096
#define D_CNT 64
#define L_CNT 3

typedef __attribute__((ext_vector_type(16))) _Float16 v16h;
typedef __attribute__((ext_vector_type(8)))  float    v8f;
typedef __attribute__((ext_vector_type(4)))  unsigned v4u;
typedef __attribute__((ext_vector_type(8)))  int      v8i;
typedef __attribute__((ext_vector_type(4)))  int      v4i;

// ---------------- deterministic hash RNG (keep prob 0.9) ----------------
__device__ __forceinline__ unsigned hash32(unsigned x) {
  x ^= x >> 16; x *= 0x7feb352dU;
  x ^= x >> 15; x *= 0x846ca68bU;
  x ^= x >> 16;
  return x;
}
#define KEEP_THR 3865470566u   // 0.9 * 2^32
__device__ __forceinline__ bool keep_rand(unsigned seed, unsigned idx) {
  return hash32(seed ^ (idx * 2654435761u)) < KEEP_THR;
}

// ---------------- TDM: DMA a 64x64 f32 tensor from global into LDS ----------------
// 6-arg builtin (upstream clang-23 lane): (v4u g0, v8i g1, v4i g2, v4i g3, v8i g4, i32 cpol)
__device__ __forceinline__ void tdm_load_64x64_f32(const float* g, unsigned lds_off) {
  unsigned long long ga = (unsigned long long)g;
  v4u g0;
  g0[0] = 1u;                                               // count=1, user mode, no gather
  g0[1] = lds_off;                                          // lds_addr [63:32]
  g0[2] = (unsigned)(ga & 0xFFFFFFFFu);                     // global_addr[31:0]
  g0[3] = (unsigned)((ga >> 32) & 0x1FFFFFFu) | (2u << 30); // global_addr[56:32], type=2
  v8i g1;
  g1[0] = (int)(2u << 16);     // workgroup_mask=0, data_size=2 (4 bytes)
  g1[1] = (int)(64u << 16);    // tensor_dim0 = 64   (bits 79:48)
  g1[2] = (int)(64u << 16);    // tensor_dim1 = 64   (bits 111:80)
  g1[3] = (int)(64u << 16);    // tile_dim0  = 64    (bits 127:112)
  g1[4] = 64;                  // tile_dim1  = 64    (bits 143:128), tile_dim2 = 0
  g1[5] = 64;                  // tensor_dim0_stride = 64 (bits 207:160)
  g1[6] = 0;
  g1[7] = 0;
  v4i z4 = {0, 0, 0, 0};       // groups 2/3 unused (2D tensor)
  v8i z8 = {0, 0, 0, 0, 0, 0, 0, 0};
  __builtin_amdgcn_tensor_load_to_lds(g0, g1, z4, z4, z8, 0);
}

// ---------------- ego = concat(user_emb, item_emb) ----------------
__global__ void k_init_ego(const float* __restrict__ ue, const float* __restrict__ ie,
                           float* __restrict__ ego) {
  long i = (long)blockIdx.x * blockDim.x + threadIdx.x;   // float4 index
  const long total4 = (long)N_CNT * D_CNT / 4;
  const long usplit = (long)U_CNT * D_CNT / 4;
  if (i >= total4) return;
  float4 v = (i < usplit) ? ((const float4*)ue)[i] : ((const float4*)ie)[i - usplit];
  ((float4*)ego)[i] = v;
}

// ---------------- zero fill (float4 granularity) ----------------
__global__ void k_zero(float4* __restrict__ p, long n4) {
  long i = (long)blockIdx.x * blockDim.x + threadIdx.x;
  if (i < n4) p[i] = make_float4(0.f, 0.f, 0.f, 0.f);
}

// ---------------- SpMM with edge dropout: side[row] += val_d * ego[col] ----------------
// Batched metadata: each lane loads a distinct edge's (val,row,col) coalesced, then the
// wave sweeps the 32 edges via shfl broadcast; per edge: 2 floats/lane gather + 2 atomics.
__global__ void k_spmm(const float* __restrict__ ego, float* __restrict__ side,
                       const float* __restrict__ val, const int* __restrict__ row,
                       const int* __restrict__ col) {
  const int lane = threadIdx.x & 31;
  long wid = (long)blockIdx.x * (blockDim.x >> 5) + (threadIdx.x >> 5);
  long nw  = (long)gridDim.x * (blockDim.x >> 5);
  for (long base = wid * 32; base < NNZ_CNT; base += nw * 32) {   // NNZ % 32 == 0
    long e_mine = base + lane;
    float v = val[e_mine];
    int   r = row[e_mine];
    int   c = col[e_mine];
    v = keep_rand(0xC0FFEE42u, (unsigned)e_mine) ? v * (1.0f / 0.9f) : 0.f;
    #pragma unroll 4
    for (int j = 0; j < 32; ++j) {
      float vj = __shfl(v, j, 32);
      int   cj = __shfl(c, j, 32);
      if (j < 31 && lane == 0)      // prefetch next edge's source row
        __builtin_prefetch(ego + (long)__shfl(c, j + 1, 32) * 64, 0, 0);
      if (vj == 0.f) continue;      // edge dropped (wave-uniform branch)
      int rj = __shfl(r, j, 32);
      float2 x = *(const float2*)(ego + (long)cj * 64 + lane * 2);
      float* dst = side + (long)rj * 64 + lane * 2;
      atomicAdd(dst,     x.x * vj);
      atomicAdd(dst + 1, x.y * vj);
    }
  }
}

// ---------------- fused dual-GEMM + bias + leakyReLU + dropout + L2 norm ----------------
// out[n] = normalize(dropout(leaky(side@Wgc + bgc + (ego*side)@Wbi + bbi)))
// block = 128 threads = 4 waves; block owns rows [16*bid, 16*bid+16), wave w owns cols [16w,16w+16)
// Weights are DMA'd into LDS by the Tensor Data Mover, then converted to transposed f16.
__launch_bounds__(128)
__global__ void k_gemm_fused(const float* __restrict__ ego, const float* __restrict__ side,
                             const float* __restrict__ Wgc, const float* __restrict__ bgc,
                             const float* __restrict__ Wbi, const float* __restrict__ bbi,
                             float* __restrict__ out, int layer) {
  __shared__ float    Wstage[2][64 * 64];   // raw f32 tiles from TDM
  __shared__ _Float16 WTgc[64 * 64];        // transposed f16: WT[col][k]
  __shared__ _Float16 WTbi[64 * 64];
  __shared__ float    tile[16 * 68];
  __shared__ float    scale[16];

  const int t = threadIdx.x;

  // --- async-tensor weight staging: one wave issues both DMAs, waits on TENSORcnt ---
  if ((t >> 5) == 0) {
    unsigned base = (unsigned)(unsigned long long)(void*)&Wstage[0][0];
    tdm_load_64x64_f32(Wgc, base);
    tdm_load_64x64_f32(Wbi, base + 64 * 64 * sizeof(float));
    __builtin_amdgcn_s_wait_tensorcnt(0);
  }
  __syncthreads();

  // convert staged f32 -> transposed f16 (B-fragments become contiguous ds loads)
  for (int i = t; i < 64 * 64; i += 128) {
    int c = i >> 6, kk = i & 63;
    WTgc[i] = (_Float16)Wstage[0][kk * 64 + c];
    WTbi[i] = (_Float16)Wstage[1][kk * 64 + c];
  }
  __syncthreads();

  const int w      = t >> 5;            // wave -> column tile
  const int lane   = t & 31;
  const int m0     = blockIdx.x * 16;   // N_CNT/16 = 18750 exact, EXEC all-ones
  const int rloc   = lane & 15;
  const int hi     = lane >> 4;
  const int colloc = (lane & 15) + 16 * w;

  v8f acc = {0.f, 0.f, 0.f, 0.f, 0.f, 0.f, 0.f, 0.f};

  #pragma unroll
  for (int s = 0; s < 2; ++s) {                  // two K=32 WMMA steps cover K=64
    const int kbA = s * 32 + hi * 8;             // A-fragment K base for this half-wave
    const float* srow = side + (long)(m0 + rloc) * 64 + kbA;
    const float* erow = ego  + (long)(m0 + rloc) * 64 + kbA;
    float ss[16], ee[16];
    *(float4*)(ss + 0)  = *(const float4*)(srow + 0);
    *(float4*)(ss + 4)  = *(const float4*)(srow + 4);
    *(float4*)(ss + 8)  = *(const float4*)(srow + 16);
    *(float4*)(ss + 12) = *(const float4*)(srow + 20);
    *(float4*)(ee + 0)  = *(const float4*)(erow + 0);
    *(float4*)(ee + 4)  = *(const float4*)(erow + 4);
    *(float4*)(ee + 8)  = *(const float4*)(erow + 16);
    *(float4*)(ee + 12) = *(const float4*)(erow + 20);
    v16h a1, a2;
    #pragma unroll
    for (int i = 0; i < 16; ++i) {
      a1[i] = (_Float16)ss[i];
      a2[i] = (_Float16)(ss[i] * ee[i]);         // (ego * side) operand
    }
    const int kbB = s * 32 + hi * 8;             // B-fragment K base
    const _Float16* pg = WTgc + colloc * 64 + kbB;
    const _Float16* pb = WTbi + colloc * 64 + kbB;
    v16h b1, b2;
    #pragma unroll
    for (int i = 0; i < 16; ++i) { b1[i] = pg[i < 8 ? i : i + 8]; b2[i] = pb[i < 8 ? i : i + 8]; }

    acc = __builtin_amdgcn_wmma_f32_16x16x32_f16(false, a1, false, b1, (short)0, acc, false, false);
    acc = __builtin_amdgcn_wmma_f32_16x16x32_f16(false, a2, false, b2, (short)0, acc, false, false);
  }

  // epilogue: bias + leaky_relu(0.2) + message dropout -> LDS tile
  const float bias = bgc[colloc] + bbi[colloc];
  #pragma unroll
  for (int i = 0; i < 8; ++i) {
    int rm = i + hi * 8;
    float x = acc[i] + bias;
    x = (x > 0.f) ? x : 0.2f * x;
    unsigned idx = (unsigned)((m0 + rm) * 64 + colloc);
    x = keep_rand(0xD00D0000u + (unsigned)(layer + 1), idx) ? x * (1.0f / 0.9f) : 0.f;
    tile[rm * 68 + colloc] = x;
  }
  __syncthreads();

  if (t < 16) {                                   // per-row L2 norm
    float sum = 0.f;
    #pragma unroll 8
    for (int j = 0; j < 64; ++j) { float v = tile[t * 68 + j]; sum += v * v; }
    scale[t] = 1.0f / fmaxf(sqrtf(sum), 1e-12f);
  }
  __syncthreads();

  #pragma unroll
  for (int i = 0; i < 8; ++i) {                   // 128 thr x 8 = 16x64 writeback
    int idx = t * 8 + i;
    int rm = idx >> 6, cc = idx & 63;
    out[(long)(m0 + rm) * 64 + cc] = tile[rm * 68 + cc] * scale[rm];
  }
}

// ---------------- per-layer BPR dot accumulation (one wave per batch element) --------
__global__ void k_dotacc(const float* __restrict__ emb, const int* __restrict__ uid,
                         const int* __restrict__ iid, const int* __restrict__ nid,
                         float* __restrict__ pos, float* __restrict__ neg) {
  int wid  = ((int)blockIdx.x * blockDim.x + (int)threadIdx.x) >> 5;
  int lane = threadIdx.x & 31;
  if (wid >= B_CNT) return;
  int u = uid[wid], ip = iid[wid], in_ = nid[wid];
  float2 ue = *(const float2*)(emb + (long)u * 64 + lane * 2);
  float2 pe = *(const float2*)(emb + ((long)U_CNT + ip) * 64 + lane * 2);
  float2 ne = *(const float2*)(emb + ((long)U_CNT + in_) * 64 + lane * 2);
  float dp = ue.x * pe.x + ue.y * pe.y;
  float dn = ue.x * ne.x + ue.y * ne.y;
  #pragma unroll
  for (int off = 16; off; off >>= 1) {
    dp += __shfl_xor(dp, off, 32);
    dn += __shfl_xor(dn, off, 32);
  }
  if (lane == 0) { pos[wid] += dp; neg[wid] += dn; }
}

// ---------------- final loss: BPR + L2 reg ----------------
__launch_bounds__(256)
__global__ void k_loss(const float* __restrict__ pos, const float* __restrict__ neg,
                       const float* __restrict__ ue, const float* __restrict__ ie,
                       const int* __restrict__ uid, const int* __restrict__ iid,
                       const int* __restrict__ nid, float* __restrict__ out) {
  __shared__ float sb[256], sr[256];
  int t = threadIdx.x;
  float accb = 0.f, accr = 0.f;
  for (int b = t; b < B_CNT; b += 256) {
    float x = pos[b] - neg[b];
    float ls = fminf(x, 0.f) - log1pf(expf(-fabsf(x)));  // stable log_sigmoid
    accb -= ls;
  }
  for (int i = t; i < B_CNT * 64; i += 256) {
    int b = i >> 6, d = i & 63;
    float a = ue[(long)uid[b] * 64 + d];
    float p = ie[(long)iid[b] * 64 + d];
    float n = ie[(long)nid[b] * 64 + d];
    accr += a * a + p * p + n * n;
  }
  sb[t] = accb; sr[t] = accr;
  __syncthreads();
  for (int s = 128; s; s >>= 1) {
    if (t < s) { sb[t] += sb[t + s]; sr[t] += sr[t + s]; }
    __syncthreads();
  }
  if (t == 0) {
    float bpr = sb[0] / (float)B_CNT;
    float reg = 1e-4f * 0.5f * sr[0] / (float)B_CNT;
    out[0] = bpr + reg;
    out[1] = bpr;
    out[2] = reg;
  }
}

// ---------------- host launcher ----------------
extern "C" void kernel_launch(void* const* d_in, const int* in_sizes, int n_in,
                              void* d_out, int out_size, void* d_ws, size_t ws_size,
                              hipStream_t stream) {
  const float* user_emb = (const float*)d_in[0];
  const float* item_emb = (const float*)d_in[1];
  const float* W_gc     = (const float*)d_in[2];
  const float* b_gc     = (const float*)d_in[3];
  const float* W_bi     = (const float*)d_in[4];
  const float* b_bi     = (const float*)d_in[5];
  const float* val      = (const float*)d_in[6];
  const int*   row      = (const int*)d_in[7];
  const int*   col      = (const int*)d_in[8];
  const int*   uid      = (const int*)d_in[9];
  const int*   iid      = (const int*)d_in[10];
  const int*   nid      = (const int*)d_in[11];
  float* out = (float*)d_out;

  char* ws = (char*)d_ws;
  const size_t embBytes = (size_t)N_CNT * D_CNT * sizeof(float);
  float* bufA    = (float*)(ws);                  // ego ping
  float* bufSide = (float*)(ws + embBytes);       // SpMM result
  float* bufB    = (float*)(ws + 2 * embBytes);   // ego pong
  float* pos     = (float*)(ws + 3 * embBytes);   // [B]
  float* neg     = pos + B_CNT;                   // [B]

  const long emb4 = (long)N_CNT * D_CNT / 4;
  const int  blk4 = (int)((emb4 + 255) / 256);

  // ego = concat(user_emb, item_emb); pos/neg = 0
  k_init_ego<<<blk4, 256, 0, stream>>>(user_emb, item_emb, bufA);
  k_zero<<<(2 * B_CNT / 4 + 255) / 256, 256, 0, stream>>>((float4*)pos, 2 * B_CNT / 4);

  // layer-0 contribution to BPR dots
  k_dotacc<<<B_CNT * 32 / 256, 256, 0, stream>>>(bufA, uid, iid, nid, pos, neg);

  float* cur = bufA;
  float* nxt = bufB;
  for (int k = 0; k < L_CNT; ++k) {
    k_zero<<<blk4, 256, 0, stream>>>((float4*)bufSide, emb4);
    k_spmm<<<8192, 256, 0, stream>>>(cur, bufSide, val, row, col);
    k_gemm_fused<<<N_CNT / 16, 128, 0, stream>>>(cur, bufSide,
                                                 W_gc + (size_t)k * D_CNT * D_CNT,
                                                 b_gc + (size_t)k * D_CNT,
                                                 W_bi + (size_t)k * D_CNT * D_CNT,
                                                 b_bi + (size_t)k * D_CNT,
                                                 nxt, k);
    k_dotacc<<<B_CNT * 32 / 256, 256, 0, stream>>>(nxt, uid, iid, nid, pos, neg);
    float* tmp = cur; cur = nxt; nxt = tmp;
  }

  k_loss<<<1, 256, 0, stream>>>(pos, neg, user_emb, item_emb, uid, iid, nid, out);
}
